// TrajectoryDecoder_74801150427829
// MI455X (gfx1250) — matrix operand order repaired and checked by reading, not compiled, not run on hardware
//
#include <hip/hip_runtime.h>
#include <hip/hip_bf16.h>

// ---------------- problem constants ----------------
constexpr int kD    = 512;
constexpr int kNH   = 8;
constexpr int kHD   = 64;
constexpr int kNL   = 6;
constexpr int kDFF  = 2048;
constexpr int kTOK  = 1024;
constexpr int kV    = kTOK + 3;      // 1027
constexpr int kPAD  = kV - 1;        // 1026
constexpr int kB    = 256;
constexpr int kSENC = 16;
constexpr int kLTGT = 62;
constexpr int kL    = kLTGT - 1;     // 61
constexpr int kM    = kB * kL;       // 15616

// ---------------- WMMA types ----------------
typedef __attribute__((ext_vector_type(16))) __bf16 v16bf;
typedef __attribute__((ext_vector_type(8)))  float  v8f;

union FragBF {
    v16bf v;
    uint4 q[2];
};

#if __has_builtin(__builtin_amdgcn_cvt_pk_bf16_f32)
typedef __attribute__((ext_vector_type(2))) __bf16 v2bf;
__device__ inline unsigned pack2bf(float a, float b) {
    v2bf r = __builtin_amdgcn_cvt_pk_bf16_f32(a, b);
    return __builtin_bit_cast(unsigned, r);
}
#else
__device__ inline unsigned short f2bf(float f) {
    unsigned u = __float_as_uint(f);
    u += 0x7FFFu + ((u >> 16) & 1u);       // round-to-nearest-even
    return (unsigned short)(u >> 16);
}
__device__ inline unsigned pack2bf(float a, float b) {
    return (unsigned)f2bf(a) | ((unsigned)f2bf(b) << 16);
}
#endif

// ---------------- tiled bf16 WMMA GEMM: C = A * B^T + bias ----------------
// A: M x K (row-major, lda). Bw: N x K (row-major, ldb). C: M x N (ldc).
// M must be a multiple of 128 (true for all call sites: 15616, 256).
// GUARDN instantiation handles ragged N (logits GEMM, N=1027).
constexpr int BM = 128, BN = 128, BK = 32;
constexpr int LDSS = 40;   // halves; 80B row pitch (16B aligned, bank-staggered)

template <bool GUARDN>
__global__ __launch_bounds__(256) void traj_gemm_bf16(
    const float* __restrict__ A, int lda,
    const float* __restrict__ Bw, int ldb,
    const float* __restrict__ bias,
    float* __restrict__ C, int ldc,
    int M, int N, int K, int relu)
{
    __shared__ __align__(16) unsigned short As[BM * LDSS];
    __shared__ __align__(16) unsigned short Bs[BN * LDSS];

    const int tid  = threadIdx.x;
    const int lane = tid & 31;
    const int w    = tid >> 5;          // 8 waves
    const int wm   = w & 1;             // 2 waves along M (64 rows each)
    const int wn   = w >> 1;            // 4 waves along N (32 cols each)
    const int half = lane >> 4;
    const int r16  = lane & 15;

    const int bm0 = blockIdx.y * BM;
    const int bn0 = blockIdx.x * BN;

    // ---- staging geometry: thread covers float4 (r0 + 32p, c4*4..) ----
    const int c4 = tid & 7;             // constant across p (256 % 8 == 0)
    const int r0 = tid >> 3;            // base row, rows r0 + 32p, p = 0..3
    const float* aptr = A  + (size_t)(bm0 + r0) * lda + c4 * 4;
    const float* bptr = Bw + (size_t)(bn0 + r0) * ldb + c4 * 4;
    bool bok[4];
#pragma unroll
    for (int p = 0; p < 4; ++p)
        bok[p] = GUARDN ? (bn0 + r0 + 32 * p < N) : true;

    unsigned short* sA = &As[r0 * LDSS + c4 * 4];
    unsigned short* sB = &Bs[r0 * LDSS + c4 * 4];

    // ---- fragment LDS base pointers (constant per thread) ----
    const unsigned short* fA = &As[(wm * 64 + r16) * LDSS + half * 8];
    const unsigned short* fB = &Bs[(wn * 32 + r16) * LDSS + half * 8];

    const v8f vzero = {0.f, 0.f, 0.f, 0.f, 0.f, 0.f, 0.f, 0.f};
    v8f acc[4][2];
#pragma unroll
    for (int mt = 0; mt < 4; ++mt)
#pragma unroll
        for (int nt = 0; nt < 2; ++nt) acc[mt][nt] = vzero;

    const float4 f4z = make_float4(0.f, 0.f, 0.f, 0.f);
    float4 ra[4], rb[4];
    // prologue: stage tile 0
#pragma unroll
    for (int p = 0; p < 4; ++p) {
        ra[p] = *(const float4*)(aptr + (size_t)(32 * p) * lda);
        rb[p] = (!GUARDN || bok[p]) ? *(const float4*)(bptr + (size_t)(32 * p) * ldb) : f4z;
    }

    for (int k0 = 0; k0 < K; k0 += BK) {
        // ---- commit staged registers to LDS ----
#pragma unroll
        for (int p = 0; p < 4; ++p) {
            uint2 pa, pb;
            pa.x = pack2bf(ra[p].x, ra[p].y);
            pa.y = pack2bf(ra[p].z, ra[p].w);
            pb.x = pack2bf(rb[p].x, rb[p].y);
            pb.y = pack2bf(rb[p].z, rb[p].w);
            *(uint2*)(sA + 32 * p * LDSS) = pa;
            *(uint2*)(sB + 32 * p * LDSS) = pb;
        }
        __syncthreads();

        // ---- stage next tile (global latency hidden behind WMMAs) ----
        if (k0 + BK < K) {
#pragma unroll
            for (int p = 0; p < 4; ++p) {
                ra[p] = *(const float4*)(aptr + (size_t)(32 * p) * lda + k0 + BK);
                rb[p] = (!GUARDN || bok[p])
                      ? *(const float4*)(bptr + (size_t)(32 * p) * ldb + k0 + BK) : f4z;
            }
        }

        // ---- fragments per ISA 16-bit A(16x32) / B(32x16) layouts ----
        FragBF af[4], bf[2];
#pragma unroll
        for (int mt = 0; mt < 4; ++mt) {
            af[mt].q[0] = *(const uint4*)(fA + mt * 16 * LDSS);       // K = 8*half..+7
            af[mt].q[1] = *(const uint4*)(fA + mt * 16 * LDSS + 16);  // K = 16+8*half..+7
        }
#pragma unroll
        for (int nt = 0; nt < 2; ++nt) {
            bf[nt].q[0] = *(const uint4*)(fB + nt * 16 * LDSS);       // K = 8*half..+7
            bf[nt].q[1] = *(const uint4*)(fB + nt * 16 * LDSS + 16);  // K = 16+8*half..+7
        }
        // ---- 8 WMMAs per wave per K-step ----
#pragma unroll
        for (int mt = 0; mt < 4; ++mt)
#pragma unroll
            for (int nt = 0; nt < 2; ++nt)
                acc[mt][nt] = __builtin_amdgcn_wmma_f32_16x16x32_bf16(
                    false, af[mt].v, false, bf[nt].v,
                    (short)0, acc[mt][nt], false, false);
        __syncthreads();
    }

    // ---- epilogue: lanes 0-15 -> M=r, N=lane ; lanes 16-31 -> M=8+r ----
#pragma unroll
    for (int mt = 0; mt < 4; ++mt)
#pragma unroll
        for (int nt = 0; nt < 2; ++nt) {
            int n = bn0 + wn * 32 + nt * 16 + r16;
            if (GUARDN && n >= N) continue;
            float bz = bias[n];
            int mb = bm0 + wm * 64 + mt * 16 + half * 8;
#pragma unroll
            for (int r = 0; r < 8; ++r) {
                float vv = acc[mt][nt][r] + bz;
                if (relu) vv = fmaxf(vv, 0.f);
                C[(size_t)(mb + r) * ldc + n] = vv;
            }
        }
}

// ---------------- embedding + point + positional ----------------
__global__ __launch_bounds__(256) void traj_embed(
    const int* __restrict__ tgt, const float* __restrict__ emb,
    const float* __restrict__ point, const float* __restrict__ pos,
    float* __restrict__ X)
{
    int m = blockIdx.x;
    int b = m / kL, l = m % kL;
    int tok = tgt[b * kLTGT + l];
    for (int d = threadIdx.x; d < kD; d += 256)
        X[(size_t)m * kD + d] = emb[(size_t)tok * kD + d]
                              + point[(size_t)b * kD + d]
                              + pos[(size_t)l * kD + d];
}

// ---------------- fused residual add + LayerNorm (in place on X) ----------------
// residual row = m / rdiv  (rdiv==1 normal, rdiv==kL broadcast over sequence)
__global__ __launch_bounds__(256) void traj_add_ln(
    float* __restrict__ X, const float* __restrict__ R, int rdiv,
    const float* __restrict__ sc, const float* __restrict__ bi)
{
    __shared__ float s1[256], s2[256];
    int m = blockIdx.x;
    int rr = m / rdiv;
    int t = threadIdx.x;
    float t0 = X[(size_t)m * kD + t]       + R[(size_t)rr * kD + t];
    float t1 = X[(size_t)m * kD + t + 256] + R[(size_t)rr * kD + t + 256];
    s1[t] = t0 + t1;
    s2[t] = t0 * t0 + t1 * t1;
    __syncthreads();
    for (int s = 128; s > 0; s >>= 1) {
        if (t < s) { s1[t] += s1[t + s]; s2[t] += s2[t + s]; }
        __syncthreads();
    }
    float mean = s1[0] * (1.f / kD);
    float var  = s2[0] * (1.f / kD) - mean * mean;
    float inv  = rsqrtf(var + 1e-5f);
    X[(size_t)m * kD + t]       = (t0 - mean) * inv * sc[t]       + bi[t];
    X[(size_t)m * kD + t + 256] = (t1 - mean) * inv * sc[t + 256] + bi[t + 256];
}

// ---------------- masked self-attention (per batch x head) ----------------
// qkv: M x 1536 rows (q | k | v). O: M x 512.
__global__ __launch_bounds__(64) void traj_attn(
    const float* __restrict__ qkv, const int* __restrict__ tgt,
    float* __restrict__ O)
{
    __shared__ float Ks[kL * kHD];
    __shared__ float Vs[kL * kHD];
    __shared__ float km[kL];
    int bh = blockIdx.x;
    int b = bh >> 3, h = bh & 7;
    int tid = threadIdx.x;

    for (int idx = tid; idx < kL * kHD; idx += 64) {
        int kk = idx >> 6, d = idx & 63;
        size_t row = (size_t)(b * kL + kk) * (3 * kD);
        Ks[idx] = qkv[row + kD     + h * kHD + d];
        Vs[idx] = qkv[row + 2 * kD + h * kHD + d];
    }
    if (tid < kL) km[tid] = (tgt[b * kLTGT + tid] == kPAD) ? -INFINITY : 0.f;
    __syncthreads();
    if (tid >= kL) return;

    int q = tid;
    float qv[kHD];
    size_t qrow = (size_t)(b * kL + q) * (3 * kD) + h * kHD;
#pragma unroll
    for (int d = 0; d < kHD; ++d) qv[d] = qkv[qrow + d];

    float sc[kL];
    float mx = -INFINITY;
    for (int k = 0; k < kL; ++k) {
        float s = 0.f;
#pragma unroll
        for (int d = 0; d < kHD; ++d) s += qv[d] * Ks[k * kHD + d];
        s = s * 0.125f + km[k];               // 1/sqrt(64)
        if (k > q) s = -INFINITY;             // causal
        sc[k] = s;
        mx = fmaxf(mx, s);
    }
    float sum = 0.f;
    for (int k = 0; k < kL; ++k) {
        float p = __expf(sc[k] - mx);
        sc[k] = p;
        sum += p;
    }
    float inv = 1.f / sum;
    size_t orow = (size_t)(b * kL + q) * kD + h * kHD;
#pragma unroll 4
    for (int d = 0; d < kHD; ++d) {
        float a = 0.f;
        for (int k = 0; k < kL; ++k) a += sc[k] * Vs[k * kHD + d];
        O[orow + d] = a * inv;
    }
}

// ---------------- host orchestration ----------------
static void launch_gemm(const float* A, int lda, const float* Bw, int ldb,
                        const float* bias, float* C, int ldc,
                        int M, int N, int K, int relu, hipStream_t s)
{
    dim3 g((N + BN - 1) / BN, (M + BM - 1) / BM), blk(256);
    if (N % BN == 0)
        traj_gemm_bf16<false><<<g, blk, 0, s>>>(A, lda, Bw, ldb, bias, C, ldc, M, N, K, relu);
    else
        traj_gemm_bf16<true><<<g, blk, 0, s>>>(A, lda, Bw, ldb, bias, C, ldc, M, N, K, relu);
}

extern "C" void kernel_launch(void* const* d_in, const int* in_sizes, int n_in,
                              void* d_out, int out_size, void* d_ws, size_t ws_size,
                              hipStream_t stream) {
    const float* encoder_out = (const float*)d_in[0];
    const float* point_out   = (const float*)d_in[1];
    const float* emb         = (const float*)d_in[2];
    const float* pos_embed   = (const float*)d_in[3];
    const float* sa_inw      = (const float*)d_in[4];
    const float* sa_inb      = (const float*)d_in[5];
    const float* sa_outw     = (const float*)d_in[6];
    const float* sa_outb     = (const float*)d_in[7];
    const float* ca_inw      = (const float*)d_in[8];
    const float* ca_inb      = (const float*)d_in[9];
    const float* ca_outw     = (const float*)d_in[10];
    const float* ca_outb     = (const float*)d_in[11];
    const float* ln1_s       = (const float*)d_in[12];
    const float* ln1_b       = (const float*)d_in[13];
    const float* ln2_s       = (const float*)d_in[14];
    const float* ln2_b       = (const float*)d_in[15];
    const float* ln3_s       = (const float*)d_in[16];
    const float* ln3_b       = (const float*)d_in[17];
    const float* ff1_w       = (const float*)d_in[18];
    const float* ff1_b       = (const float*)d_in[19];
    const float* ff2_w       = (const float*)d_in[20];
    const float* ff2_b       = (const float*)d_in[21];
    const float* out_w       = (const float*)d_in[22];
    const float* out_b       = (const float*)d_in[23];
    const int*   tgt         = (const int*)d_in[24];
    float* out = (float*)d_out;

    // workspace: X (M*D) | T1 (M*DFF) | T2 (M*D)
    float* X  = (float*)d_ws;
    float* T1 = X + (size_t)kM * kD;
    float* T2 = T1 + (size_t)kM * kDFF;
    float* O    = T1 + (size_t)kM * (3 * kD);   // attn out lives in T1 tail (DFF > 3D)
    float* VMEM = T2;                            // B x D
    float* CO   = T2 + (size_t)kB * kD;          // B x D

    // x = emb[tokens] + point_out + pos_embed
    traj_embed<<<kM, 256, 0, stream>>>(tgt, emb, point_out, pos_embed, X);

    for (int i = 0; i < kNL; ++i) {
        const size_t o3DD = (size_t)i * 3 * kD * kD;
        const size_t oDD  = (size_t)i * kD * kD;
        // ---- self attention ----
        launch_gemm(X, kD, sa_inw + o3DD, kD, sa_inb + (size_t)i * 3 * kD,
                    T1, 3 * kD, kM, 3 * kD, kD, 0, stream);                 // qkv
        traj_attn<<<kB * kNH, 64, 0, stream>>>(T1, tgt, O);
        launch_gemm(O, kD, sa_outw + oDD, kD, sa_outb + (size_t)i * kD,
                    T2, kD, kM, kD, kD, 0, stream);                         // out proj
        traj_add_ln<<<kM, 256, 0, stream>>>(X, T2, 1,
                    ln1_s + (size_t)i * kD, ln1_b + (size_t)i * kD);
        // ---- cross attention: Tk==1 => softmax==1 => output = proj(V(mem)) broadcast ----
        launch_gemm(encoder_out, kSENC * kD,
                    ca_inw + o3DD + (size_t)2 * kD * kD, kD,
                    ca_inb + (size_t)i * 3 * kD + 2 * kD,
                    VMEM, kD, kB, kD, kD, 0, stream);                       // v = mem @ wv^T
        launch_gemm(VMEM, kD, ca_outw + oDD, kD, ca_outb + (size_t)i * kD,
                    CO, kD, kB, kD, kD, 0, stream);                         // co = v @ outw^T
        traj_add_ln<<<kM, 256, 0, stream>>>(X, CO, kL,
                    ln2_s + (size_t)i * kD, ln2_b + (size_t)i * kD);
        // ---- FFN ----
        launch_gemm(X, kD, ff1_w + (size_t)i * kDFF * kD, kD,
                    ff1_b + (size_t)i * kDFF, T1, kDFF, kM, kDFF, kD, 1, stream);
        launch_gemm(T1, kDFF, ff2_w + (size_t)i * kD * kDFF, kDFF,
                    ff2_b + (size_t)i * kD, T2, kD, kM, kD, kDFF, 0, stream);
        traj_add_ln<<<kM, 256, 0, stream>>>(X, T2, 1,
                    ln3_s + (size_t)i * kD, ln3_b + (size_t)i * kD);
    }
    // logits
    launch_gemm(X, kD, out_w, kD, out_b, out, kV, kM, kV, kD, 0, stream);
}
